// LSTMModel_30081950941244
// MI455X (gfx1250) — matrix operand order repaired and checked by reading, not compiled, not run on hardware
//
#include <hip/hip_runtime.h>

typedef __attribute__((ext_vector_type(16))) _Float16 v16h;
typedef __attribute__((ext_vector_type(8)))  _Float16 v8h;
typedef __attribute__((ext_vector_type(4)))  _Float16 v4h;
typedef __attribute__((ext_vector_type(8)))  float    v8f;

#define INPUT_D 467
#define HID     128
#define GATES   512     // 4*HID
#define SEQ_T   512
#define BATCH   256
#define AK2      40     // LDS row stride (f16) for GEMM staging: 80B, 16B aligned
#define H_STRIDE 136    // LDS row stride (f16) for h buffers: 272B, 16B aligned

__device__ __forceinline__ v8f wmma16(v16h a, v16h b, v8f c) {
  return __builtin_amdgcn_wmma_f32_16x16x32_f16(false, a, false, b, (short)0, c,
                                                false, false);
}

__device__ __forceinline__ float fsigmoid(float x) {
  return __builtin_amdgcn_rcpf(1.0f + __expf(-x));
}
__device__ __forceinline__ float ftanh(float x) {
  return 2.0f * fsigmoid(2.0f * x) - 1.0f;
}

// ---- A-matrix fragment (16 rows x 32 K) from f16 LDS tile ----------------
// Contiguous K order used here (matches bfrag_f16): lane (r = l&15) holds K = hi..hi+15, hi=(l>>4)*16
__device__ __forceinline__ v16h afrag_f16(const _Float16* base, int stride, int lane) {
  const int r  = lane & 15;
  const int hi = (lane >> 4) << 4;
  const _Float16* p = base + r * stride + hi;
  v8h lo = *(const v8h*)p;          // 16B aligned
  v8h hv = *(const v8h*)(p + 8);    // 16B aligned
  return __builtin_shufflevector(lo, hv, 0, 1, 2, 3, 4, 5, 6, 7,
                                 8, 9, 10, 11, 12, 13, 14, 15);
}

// ---- B-matrix fragment (32 K x 16 cols) from f16 LDS tile ----------------
// Contiguous K order used here (matches afrag_f16): lane = col n (l&15); lanes 0-15 hold K=0..15, lanes 16-31 K=16..31
__device__ __forceinline__ v16h bfrag_f16(const _Float16* base, int stride, int lane) {
  const int n  = lane & 15;
  const int k0 = (lane >> 4) << 4;
  const _Float16* p = base + n * stride + k0;
  v8h lo = *(const v8h*)p;
  v8h hv = *(const v8h*)(p + 8);
  return __builtin_shufflevector(lo, hv, 0, 1, 2, 3, 4, 5, 6, 7,
                                 8, 9, 10, 11, 12, 13, 14, 15);
}

// ===========================================================================
// Kernel A: layer-0 input projection + biases, output in per-lane-packed
// WMMA C-fragment layout:
//   xp[(((b_tile*512 + t)*32 + tile_n)*32 + lane)*8 + v]  (f16)
// WG = 16 batch rows x 8 timesteps x 64 gate cols; wave w owns timestep t0+w.
// LDS tiles staged as f16 (converted once at staging time).
// ===========================================================================
__global__ __launch_bounds__(256) void xproj_gemm(
    const float* __restrict__ x, const float* __restrict__ Wih0,
    const float* __restrict__ bih0, const float* __restrict__ bhh0,
    _Float16* __restrict__ xp) {
  __shared__ _Float16 Ah[128 * AK2];   // rows ordered (tslot*16 + b_r)
  __shared__ _Float16 Bh[64 * AK2];
  const int tid = threadIdx.x, lane = tid & 31, w = tid >> 5;
  const int t0 = blockIdx.x * 8;          // 64 t-tiles
  const int b0 = blockIdx.y * 16;         // 16 b-tiles
  const int n0 = blockIdx.z * 64;         // 8  n-tiles

  v8f acc[4];
#pragma unroll
  for (int j = 0; j < 4; ++j) acc[j] = (v8f){0, 0, 0, 0, 0, 0, 0, 0};

  for (int kc = 0; kc < 15; ++kc) {       // ceil(467/32) chunks, zero-padded tail
    const int k0 = kc * 32;
    __syncthreads();
    // stage A: 128 rows (16 batch x 8 t) x 32 K, fp32 -> f16, 4-elem units
    for (int u = tid; u < 1024; u += 256) {
      const int row = u >> 3, c4 = (u & 7) << 2;
      const int tslot = row >> 4, b_r = row & 15;
      const float* src =
          x + ((size_t)(b0 + b_r) * SEQ_T + (t0 + tslot)) * INPUT_D + (k0 + c4);
      float f0 = (k0 + c4 + 0 < INPUT_D) ? src[0] : 0.0f;
      float f1 = (k0 + c4 + 1 < INPUT_D) ? src[1] : 0.0f;
      float f2 = (k0 + c4 + 2 < INPUT_D) ? src[2] : 0.0f;
      float f3 = (k0 + c4 + 3 < INPUT_D) ? src[3] : 0.0f;
      auto h01 = __builtin_amdgcn_cvt_pkrtz(f0, f1);
      auto h23 = __builtin_amdgcn_cvt_pkrtz(f2, f3);
      v4h q; q[0] = h01[0]; q[1] = h01[1]; q[2] = h23[0]; q[3] = h23[1];
      *(v4h*)&Ah[row * AK2 + c4] = q;
    }
    // stage B: 64 W-rows x 32 K, fp32 -> f16
    for (int u = tid; u < 512; u += 256) {
      const int row = u >> 3, c4 = (u & 7) << 2;
      const float* src = Wih0 + (size_t)(n0 + row) * INPUT_D + (k0 + c4);
      float f0 = (k0 + c4 + 0 < INPUT_D) ? src[0] : 0.0f;
      float f1 = (k0 + c4 + 1 < INPUT_D) ? src[1] : 0.0f;
      float f2 = (k0 + c4 + 2 < INPUT_D) ? src[2] : 0.0f;
      float f3 = (k0 + c4 + 3 < INPUT_D) ? src[3] : 0.0f;
      auto h01 = __builtin_amdgcn_cvt_pkrtz(f0, f1);
      auto h23 = __builtin_amdgcn_cvt_pkrtz(f2, f3);
      v4h q; q[0] = h01[0]; q[1] = h01[1]; q[2] = h23[0]; q[3] = h23[1];
      *(v4h*)&Bh[row * AK2 + c4] = q;
    }
    __syncthreads();
    v16h a = afrag_f16(&Ah[(w * 16) * AK2], AK2, lane);
#pragma unroll
    for (int j = 0; j < 4; ++j) {
      v16h b = bfrag_f16(&Bh[(j * 16) * AK2], AK2, lane);
      acc[j] = wmma16(a, b, acc[j]);
    }
  }
  // epilogue: add bias, pack per-lane, one v8h store per tile
  const int t = t0 + w;
#pragma unroll
  for (int j = 0; j < 4; ++j) {
    const int ng = n0 + j * 16 + (lane & 15);
    const float bias = bih0[ng] + bhh0[ng];
    v8h hv;
#pragma unroll
    for (int p = 0; p < 4; ++p) {
      auto h = __builtin_amdgcn_cvt_pkrtz(acc[j][2 * p] + bias,
                                          acc[j][2 * p + 1] + bias);
      hv[2 * p] = h[0]; hv[2 * p + 1] = h[1];
    }
    const size_t tile_n = (size_t)((n0 >> 4) + j);
    _Float16* dst = xp + ((((size_t)blockIdx.y * SEQ_T + t) * 32 + tile_n) * 32 +
                          lane) * 8;
    *(v8h*)dst = hv;
  }
}

// ===========================================================================
// Kernel C: pack W_hh0 / W_ih1 / W_hh1 ([512,128] f32, row = gate col n) into
// per-lane WMMA B-fragment f16 blocks: block (n_tile 0..31, kk 0..3) holds
// 32 lanes x 16 f16: lane l -> col n = n_tile*16+(l&15), K = kk*32+(l>>4)*16+e
// ===========================================================================
__global__ __launch_bounds__(256) void pack_weights(
    const float* __restrict__ Whh0, const float* __restrict__ Wih1,
    const float* __restrict__ Whh1, _Float16* __restrict__ pk) {
  const int id = blockIdx.x * 256 + threadIdx.x;
  if (id >= 3 * 32 * 4 * 32) return;
  const int lane = id & 31;
  const int kk = (id >> 5) & 3;
  const int nt = (id >> 7) & 31;
  const int m  = id >> 12;
  const float* W = (m == 0) ? Whh0 : (m == 1) ? Wih1 : Whh1;
  const int n  = nt * 16 + (lane & 15);
  const int kb = kk * 32 + ((lane >> 4) << 4);
  _Float16* dst = pk + (size_t)m * (GATES * HID) + (size_t)(nt * 4 + kk) * 512 + lane * 16;
#pragma unroll
  for (int e = 0; e < 16; ++e)
    dst[e] = (_Float16)W[(size_t)n * HID + kb + e];
}

// ===========================================================================
// Kernel B: fused 2-layer recurrence. 16 WGs x 256 thr; WG = 16 batch rows.
// Wave w owns hidden units [16w,16w+16): gate tiles at n = g*128 + 16w.
// h0/h1 ping-pong in LDS; W_hh0 + W_ih1 LDS-resident (256KB of 320KB/WGP),
// W_hh1 streamed from L2 each step.
// ===========================================================================
__global__ __launch_bounds__(256) void lstm_recurrent(
    const _Float16* __restrict__ xp, const _Float16* __restrict__ pk,
    const float* __restrict__ bih1, const float* __restrict__ bhh1,
    const float* __restrict__ Wfc, const float* __restrict__ bfc,
    float* __restrict__ out) {
  __shared__ _Float16 wlds[2 * GATES * HID];          // 256 KB: matrices 0,1
  __shared__ _Float16 h0buf[2][16 * H_STRIDE];
  __shared__ _Float16 h1buf[2][16 * H_STRIDE];
  const int tid = threadIdx.x, lane = tid & 31, w = tid >> 5;
  const int b0 = blockIdx.x * 16;
  const int nlo = lane & 15, rsel = (lane >> 4) << 3;

  // one-time: stage packed W_hh0 + W_ih1 into LDS (v8h units)
  for (int i = tid; i < (2 * GATES * HID) / 8; i += 256)
    *(v8h*)&wlds[i * 8] = *(const v8h*)&pk[i * 8];

  for (int i = tid; i < 16 * H_STRIDE; i += 256) {
    h0buf[0][i] = (_Float16)0.0f; h0buf[1][i] = (_Float16)0.0f;
    h1buf[0][i] = (_Float16)0.0f; h1buf[1][i] = (_Float16)0.0f;
  }

  float bias1[4];
#pragma unroll
  for (int g = 0; g < 4; ++g) {
    const int n = g * 128 + w * 16 + nlo;
    bias1[g] = bih1[n] + bhh1[n];
  }
  // fragment block offsets for this wave's gate tiles (n_tile = g*8+w)
  int wofs[4];
#pragma unroll
  for (int g = 0; g < 4; ++g) wofs[g] = ((g * 8 + w) * 4) * 512 + lane * 16;
  const _Float16* pk2 = pk + (size_t)2 * (GATES * HID);   // W_hh1 (global)

  // per-lane base into packed xp for this (b_tile, lane)
  const _Float16* xpb = xp + (size_t)blockIdx.x * SEQ_T * 32 * 256 + (size_t)lane * 8;

  v8f c0 = (v8f){0, 0, 0, 0, 0, 0, 0, 0};
  v8f c1 = (v8f){0, 0, 0, 0, 0, 0, 0, 0};
  __syncthreads();

  for (int t = 0; t < SEQ_T; ++t) {
    const int cur = t & 1, nxt = cur ^ 1;
    // ---------------- layer 0 ----------------
    v8f g0[4];
#pragma unroll
    for (int g = 0; g < 4; ++g) {
      const v8h hv = *(const v8h*)(xpb + ((size_t)t * 32 + (g * 8 + w)) * 256);
#pragma unroll
      for (int v = 0; v < 8; ++v) g0[g][v] = (float)hv[v];
    }
    // prefetch next timestep's xp tiles (global_prefetch_b8)
#pragma unroll
    for (int g = 0; g < 4; ++g)
      __builtin_prefetch(
          (const void*)(xpb + ((size_t)(t + 1) * 32 + (g * 8 + w)) * 256), 0, 1);
#pragma unroll
    for (int kk = 0; kk < 4; ++kk) {
      v16h a = afrag_f16(&h0buf[cur][kk * 32], H_STRIDE, lane);
#pragma unroll
      for (int g = 0; g < 4; ++g) {
        v16h bb = *(const v16h*)&wlds[wofs[g] + kk * 512];        // LDS W_hh0
        g0[g] = wmma16(a, bb, g0[g]);
      }
    }
#pragma unroll
    for (int v = 0; v < 8; ++v) {
      const float iv = fsigmoid(g0[0][v]);
      const float fv = fsigmoid(g0[1][v]);
      const float gv = ftanh(g0[2][v]);
      const float ov = fsigmoid(g0[3][v]);
      const float cv = fv * c0[v] + iv * gv;
      c0[v] = cv;
      h0buf[nxt][(v + rsel) * H_STRIDE + w * 16 + nlo] = (_Float16)(ov * ftanh(cv));
    }
    __syncthreads();
    // ---------------- layer 1 ----------------
    v8f g1[4];
#pragma unroll
    for (int g = 0; g < 4; ++g)
#pragma unroll
      for (int v = 0; v < 8; ++v) g1[g][v] = bias1[g];
#pragma unroll
    for (int kk = 0; kk < 4; ++kk) {
      v16h a = afrag_f16(&h0buf[nxt][kk * 32], H_STRIDE, lane);   // new h0
#pragma unroll
      for (int g = 0; g < 4; ++g) {
        v16h bb = *(const v16h*)&wlds[GATES * HID + wofs[g] + kk * 512];  // LDS W_ih1
        g1[g] = wmma16(a, bb, g1[g]);
      }
    }
#pragma unroll
    for (int kk = 0; kk < 4; ++kk) {
      v16h a = afrag_f16(&h1buf[cur][kk * 32], H_STRIDE, lane);
#pragma unroll
      for (int g = 0; g < 4; ++g) {
        v16h bb = *(const v16h*)(pk2 + wofs[g] + kk * 512);        // global W_hh1
        g1[g] = wmma16(a, bb, g1[g]);
      }
    }
#pragma unroll
    for (int v = 0; v < 8; ++v) {
      const float iv = fsigmoid(g1[0][v]);
      const float fv = fsigmoid(g1[1][v]);
      const float gv = ftanh(g1[2][v]);
      const float ov = fsigmoid(g1[3][v]);
      const float cv = fv * c1[v] + iv * gv;
      c1[v] = cv;
      h1buf[nxt][(v + rsel) * H_STRIDE + w * 16 + nlo] = (_Float16)(ov * ftanh(cv));
    }
    __syncthreads();
  }
  // final: h2[T-1] in h1buf[0] (t=511 -> nxt=0); out = sigmoid(h @ Wfc^T + bfc)
  if (tid < 16) {
    float s = bfc[0];
    const _Float16* hr = &h1buf[0][tid * H_STRIDE];
#pragma unroll 8
    for (int k = 0; k < HID; ++k) s += (float)hr[k] * Wfc[k];
    out[b0 + tid] = fsigmoid(s);
  }
}

extern "C" void kernel_launch(void* const* d_in, const int* in_sizes, int n_in,
                              void* d_out, int out_size, void* d_ws, size_t ws_size,
                              hipStream_t stream) {
  const float* x    = (const float*)d_in[0];
  const float* Wih0 = (const float*)d_in[1];
  const float* Whh0 = (const float*)d_in[2];
  const float* bih0 = (const float*)d_in[3];
  const float* bhh0 = (const float*)d_in[4];
  const float* Wih1 = (const float*)d_in[5];
  const float* Whh1 = (const float*)d_in[6];
  const float* bih1 = (const float*)d_in[7];
  const float* bhh1 = (const float*)d_in[8];
  const float* Wfc  = (const float*)d_in[9];
  const float* bfc  = (const float*)d_in[10];
  float* out = (float*)d_out;

  _Float16* xp = (_Float16*)d_ws;   // 16 btiles * 512 t * 32 tiles * 256 f16
  _Float16* pk = (_Float16*)((char*)d_ws + (size_t)BATCH * SEQ_T * GATES * 2);

  xproj_gemm<<<dim3(SEQ_T / 8, BATCH / 16, GATES / 64), 256, 0, stream>>>(
      x, Wih0, bih0, bhh0, xp);
  pack_weights<<<48, 256, 0, stream>>>(Whh0, Wih1, Whh1, pk);
  lstm_recurrent<<<BATCH / 16, 256, 0, stream>>>(xp, pk, bih1, bhh1, Wfc, bfc, out);
}
